// ReplicatedLinearWithLoRA_43628277793189
// MI455X (gfx1250) — compile-verified, matrix operand order checked by reading, not run on hardware
//
#include <hip/hip_runtime.h>
#include <hip/hip_bf16.h>

// ---------------------------------------------------------------------------
// ReplicatedLinearWithLoRA for MI455X (gfx1250, wave32, WMMA, async-to-LDS)
//   out = x @ W^T + base_bias + onehot@bias_stacked + B[idx] @ (A[idx] @ x)
//
// Fast path (needs ~96.5 MB workspace):
//   1) shrink:  s[t,r] = x[t,:] . lora_a[idx[t],r,:]            (scalar, LDS)
//   2) prepass: x,W f32 -> bf16 in workspace (halves all tile traffic)
//   3) GEMM:    128x128x64 tiles, double-buffered LDS fed by
//               global_load_async_to_lds_b128 (+ s_wait_asynccnt), 16x
//               v_wmma_f32_16x16x32_bf16 per wave per K-step, fused
//               bias + rank-16 LoRA-expand epilogue.
// Fallback (small ws): round-1 kernel converting f32->bf16 inside the GEMM.
// ---------------------------------------------------------------------------

typedef __attribute__((ext_vector_type(16))) __bf16 v16bf;
typedef __attribute__((ext_vector_type(8)))  __bf16 v8bf;
typedef __attribute__((ext_vector_type(4)))  __bf16 v4bf;
typedef __attribute__((ext_vector_type(8)))  float  v8f;

#define T_TOK  8192
#define D_IN   4096
#define D_OUTD 4096
#define R_RANK 16

// ---------------------------------------------------------------------------
// Kernel 1: LoRA shrink  s[t, r] = dot(x[t, :], lora_a[idx[t], r, :])
// ---------------------------------------------------------------------------
__global__ __launch_bounds__(256)
void lora_shrink_kernel(const float* __restrict__ x,
                        const float* __restrict__ lora_a,
                        const int*   __restrict__ indices,
                        float*       __restrict__ s)
{
    __shared__ float xrow[D_IN];
    const int t   = blockIdx.x;
    const int tid = threadIdx.x;
    const int idx = indices[t];

    if (idx < 0) {                       // uniform across block: safe early-out
        if (tid < R_RANK) s[(size_t)t * R_RANK + tid] = 0.0f;
        return;
    }

    const float* xp = x + (size_t)t * D_IN;
    for (int d = tid; d < D_IN; d += 256) xrow[d] = xp[d];
    __syncthreads();

    const int wave = tid >> 5;
    const int lane = tid & 31;
    const int r0 = wave * 2, r1 = r0 + 1;
    const float* a0 = lora_a + ((size_t)idx * R_RANK + r0) * D_IN;
    const float* a1 = lora_a + ((size_t)idx * R_RANK + r1) * D_IN;

    float acc0 = 0.f, acc1 = 0.f;
    for (int d = lane; d < D_IN; d += 32) {
        const float xv = xrow[d];
        acc0 += xv * a0[d];
        acc1 += xv * a1[d];
    }
    #pragma unroll
    for (int off = 16; off >= 1; off >>= 1) {
        acc0 += __shfl_xor(acc0, off, 32);
        acc1 += __shfl_xor(acc1, off, 32);
    }
    if (lane == 0) {
        s[(size_t)t * R_RANK + r0] = acc0;
        s[(size_t)t * R_RANK + r1] = acc1;
    }
}

// ---------------------------------------------------------------------------
// Kernel 2: f32 -> bf16 conversion prepass (x and W), 4 elements per thread
// ---------------------------------------------------------------------------
__global__ __launch_bounds__(256)
void cvt_f32_to_bf16_kernel(const float* __restrict__ src,
                            __bf16*      __restrict__ dst, long n4)
{
    const long i = (long)blockIdx.x * 256 + threadIdx.x;
    if (i >= n4) return;
    const float4 v = ((const float4*)src)[i];
    v4bf o;
    o[0] = (__bf16)v.x; o[1] = (__bf16)v.y;
    o[2] = (__bf16)v.z; o[3] = (__bf16)v.w;
    ((v4bf*)dst)[i] = o;
}

// ---------------------------------------------------------------------------
// Kernel 3 (fast): bf16 WMMA GEMM, async-to-LDS double-buffered tile feed.
// Block 128x128, K-step 64 (two wmma k-slices). 8 waves in 2(M)x4(N) grid,
// each wave owns a 64x32 stripe = 4x2 tiles of 16x16.
// ---------------------------------------------------------------------------
#define BM 128
#define BN 128
#define BK 64
#define LDT2 (BK + 8)                    // 72 elems -> 144 B stride (16B mult)
#define A_ELEMS (BM * LDT2)              // 9216
#define B_ELEMS (BN * LDT2)              // 9216
#define BUF_ELEMS (A_ELEMS + B_ELEMS)    // 18432 elems per buffer (36 KB)

__global__ __launch_bounds__(256)
void lora_gemm_async_kernel(const __bf16* __restrict__ xb,
                            const __bf16* __restrict__ wb,
                            const float* __restrict__ base_bias,
                            const float* __restrict__ lora_b,
                            const float* __restrict__ bias_stacked,
                            const int*   __restrict__ indices,
                            const float* __restrict__ s,
                            float*       __restrict__ out)
{
    __shared__ __bf16 smem[2 * BUF_ELEMS];          // 72 KB double buffer

    const int tid   = threadIdx.x;
    const int bn    = blockIdx.x;
    const int bm    = blockIdx.y;
    const int wave  = tid >> 5;
    const int lane  = tid & 31;
    const int wm    = wave >> 2;         // 0..1
    const int wn    = wave & 3;          // 0..3
    const int laneM = lane & 15;
    const int laneH = lane >> 4;
    const int kb    = laneH * 8;

    const unsigned smem_base = (unsigned)(size_t)&smem[0];
    const __bf16* xTile = xb + ((size_t)bm * BM) * D_IN;
    const __bf16* wTile = wb + ((size_t)bn * BN) * D_IN;

    // Issue one 64-K tile-pair of async LDS loads (no VGPR staging):
    // 128 rows x 8 chunks(16B) per matrix = 1024 chunks; 4 per thread each.
    auto issue_tile = [&](int bufsel, int kk) {
        const unsigned bufbase = smem_base + (unsigned)(bufsel * BUF_ELEMS * 2);
        #pragma unroll
        for (int i = 0; i < 4; ++i) {
            const int q   = tid + i * 256;       // 0..1023
            const int row = q >> 3;              // 0..127
            const int ck  = q & 7;               // 16B chunk in the 64-K row
            const unsigned la = bufbase + (unsigned)((row * LDT2 + ck * 8) * 2);
            const __bf16*  ga = xTile + (size_t)row * D_IN + kk + ck * 8;
            asm volatile("global_load_async_to_lds_b128 %0, %1, off"
                         :: "v"(la), "v"(ga) : "memory");
            const unsigned lb = bufbase +
                (unsigned)((A_ELEMS + row * LDT2 + ck * 8) * 2);
            const __bf16*  gb = wTile + (size_t)row * D_IN + kk + ck * 8;
            asm volatile("global_load_async_to_lds_b128 %0, %1, off"
                         :: "v"(lb), "v"(gb) : "memory");
        }
    };

    v8f acc[4][2];
    #pragma unroll
    for (int mt = 0; mt < 4; ++mt)
        #pragma unroll
        for (int nt = 0; nt < 2; ++nt)
            acc[mt][nt] = (v8f){0.f,0.f,0.f,0.f,0.f,0.f,0.f,0.f};

    issue_tile(0, 0);

    const int NIT = D_IN / BK;                      // 64 iterations
    for (int it = 0; it < NIT; ++it) {
        const int cur = it & 1;
        // my async writes into buf[cur] complete:
        asm volatile("s_wait_asynccnt 0x0" ::: "memory");
        // everyone's writes complete; everyone's reads of buf[cur^1] from the
        // previous iteration completed before their barrier-signal:
        __syncthreads();
        if (it + 1 < NIT) issue_tile(cur ^ 1, (it + 1) * BK);

        const __bf16* Ab = &smem[cur * BUF_ELEMS];
        const __bf16* Bb = &smem[cur * BUF_ELEMS + A_ELEMS];

        #pragma unroll
        for (int ks = 0; ks < 2; ++ks) {            // two 32-K wmma slices
            const int ko = ks * 32 + kb;
            v16bf afrag[4], bfrag[2];
            #pragma unroll
            for (int mt = 0; mt < 4; ++mt) {
                const __bf16* p = &Ab[(wm * 64 + mt * 16 + laneM) * LDT2 + ko];
                const v8bf lo = *(const v8bf*)p;         // K = ko .. ko+7
                const v8bf hi = *(const v8bf*)(p + 16);  // K = ko+16 .. ko+23
                afrag[mt] = __builtin_shufflevector(lo, hi,
                    0,1,2,3,4,5,6,7,8,9,10,11,12,13,14,15);
            }
            #pragma unroll
            for (int nt = 0; nt < 2; ++nt) {
                const __bf16* p = &Bb[(wn * 32 + nt * 16 + laneM) * LDT2 + ko];
                const v8bf lo = *(const v8bf*)p;
                const v8bf hi = *(const v8bf*)(p + 16);
                bfrag[nt] = __builtin_shufflevector(lo, hi,
                    0,1,2,3,4,5,6,7,8,9,10,11,12,13,14,15);
            }
            #pragma unroll
            for (int mt = 0; mt < 4; ++mt)
                #pragma unroll
                for (int nt = 0; nt < 2; ++nt)
                    acc[mt][nt] = __builtin_amdgcn_wmma_f32_16x16x32_bf16(
                        false, afrag[mt], false, bfrag[nt],
                        (short)0, acc[mt][nt], false, false);
        }
    }

    // --- epilogue: + base_bias + bias_stacked[idx] + s[t] . lora_b[idx] ----
    const int mBase = bm * BM + wm * 64;
    const int nBase = bn * BN + wn * 32;
    #pragma unroll
    for (int mt = 0; mt < 4; ++mt) {
        #pragma unroll
        for (int v = 0; v < 8; ++v) {
            const int m   = mBase + mt * 16 + laneH * 8 + v;
            const int idx = indices[m];
            const float* srow = s + (size_t)m * R_RANK;
            #pragma unroll
            for (int nt = 0; nt < 2; ++nt) {
                const int n = nBase + nt * 16 + laneM;
                float val = acc[mt][nt][v] + base_bias[n];
                if (idx >= 0) {
                    val += bias_stacked[(size_t)idx * D_OUTD + n];
                    const float* bcol =
                        lora_b + ((size_t)idx * D_OUTD + n) * R_RANK;
                    float d = 0.f;
                    #pragma unroll
                    for (int r = 0; r < R_RANK; ++r) d += srow[r] * bcol[r];
                    val += d;
                }
                out[(size_t)m * D_OUTD + n] = val;
            }
        }
    }
}

// ---------------------------------------------------------------------------
// Fallback GEMM (round-1): f32->bf16 conversion inside the kernel,
// single-buffered LDS. Used only when workspace can't hold bf16 copies.
// ---------------------------------------------------------------------------
#define FBK 32
#define FLDT (FBK + 8)

__global__ __launch_bounds__(256)
void lora_gemm_fallback_kernel(const float* __restrict__ x,
                               const float* __restrict__ W,
                               const float* __restrict__ base_bias,
                               const float* __restrict__ lora_b,
                               const float* __restrict__ bias_stacked,
                               const int*   __restrict__ indices,
                               const float* __restrict__ s,
                               float*       __restrict__ out)
{
    __shared__ __bf16 As[BM * FLDT];
    __shared__ __bf16 Bs[BN * FLDT];

    const int tid   = threadIdx.x;
    const int bn    = blockIdx.x;
    const int bm    = blockIdx.y;
    const int wave  = tid >> 5;
    const int lane  = tid & 31;
    const int wm    = wave >> 2;
    const int wn    = wave & 3;
    const int laneM = lane & 15;
    const int laneH = lane >> 4;
    const int kb    = laneH * 8;

    v8f acc[4][2];
    #pragma unroll
    for (int mt = 0; mt < 4; ++mt)
        #pragma unroll
        for (int nt = 0; nt < 2; ++nt)
            acc[mt][nt] = (v8f){0.f,0.f,0.f,0.f,0.f,0.f,0.f,0.f};

    const float* xBase = x + ((size_t)bm * BM) * D_IN;
    const float* wBase = W + ((size_t)bn * BN) * D_IN;

    for (int kk = 0; kk < D_IN; kk += FBK) {
        __syncthreads();
        #pragma unroll
        for (int i = 0; i < 4; ++i) {
            const int q   = tid + i * 256;
            const int row = q >> 3;
            const int kc  = q & 7;
            const float4 va = *(const float4*)(xBase + (size_t)row * D_IN + kk + kc * 4);
            __bf16* da = &As[row * FLDT + kc * 4];
            da[0] = (__bf16)va.x; da[1] = (__bf16)va.y;
            da[2] = (__bf16)va.z; da[3] = (__bf16)va.w;
            const float4 vb = *(const float4*)(wBase + (size_t)row * D_IN + kk + kc * 4);
            __bf16* db = &Bs[row * FLDT + kc * 4];
            db[0] = (__bf16)vb.x; db[1] = (__bf16)vb.y;
            db[2] = (__bf16)vb.z; db[3] = (__bf16)vb.w;
        }
        __syncthreads();

        v16bf afrag[4], bfrag[2];
        #pragma unroll
        for (int mt = 0; mt < 4; ++mt) {
            const __bf16* p = &As[(wm * 64 + mt * 16 + laneM) * FLDT + kb];
            const v8bf lo = *(const v8bf*)p;
            const v8bf hi = *(const v8bf*)(p + 16);
            afrag[mt] = __builtin_shufflevector(lo, hi,
                0,1,2,3,4,5,6,7,8,9,10,11,12,13,14,15);
        }
        #pragma unroll
        for (int nt = 0; nt < 2; ++nt) {
            const __bf16* p = &Bs[(wn * 32 + nt * 16 + laneM) * FLDT + kb];
            const v8bf lo = *(const v8bf*)p;
            const v8bf hi = *(const v8bf*)(p + 16);
            bfrag[nt] = __builtin_shufflevector(lo, hi,
                0,1,2,3,4,5,6,7,8,9,10,11,12,13,14,15);
        }
        #pragma unroll
        for (int mt = 0; mt < 4; ++mt)
            #pragma unroll
            for (int nt = 0; nt < 2; ++nt)
                acc[mt][nt] = __builtin_amdgcn_wmma_f32_16x16x32_bf16(
                    false, afrag[mt], false, bfrag[nt],
                    (short)0, acc[mt][nt], false, false);
    }

    const int mBase = bm * BM + wm * 64;
    const int nBase = bn * BN + wn * 32;
    #pragma unroll
    for (int mt = 0; mt < 4; ++mt) {
        #pragma unroll
        for (int v = 0; v < 8; ++v) {
            const int m   = mBase + mt * 16 + laneH * 8 + v;
            const int idx = indices[m];
            const float* srow = s + (size_t)m * R_RANK;
            #pragma unroll
            for (int nt = 0; nt < 2; ++nt) {
                const int n = nBase + nt * 16 + laneM;
                float val = acc[mt][nt][v] + base_bias[n];
                if (idx >= 0) {
                    val += bias_stacked[(size_t)idx * D_OUTD + n];
                    const float* bcol =
                        lora_b + ((size_t)idx * D_OUTD + n) * R_RANK;
                    float d = 0.f;
                    #pragma unroll
                    for (int r = 0; r < R_RANK; ++r) d += srow[r] * bcol[r];
                    val += d;
                }
                out[(size_t)m * D_OUTD + n] = val;
            }
        }
    }
}

// ---------------------------------------------------------------------------
extern "C" void kernel_launch(void* const* d_in, const int* in_sizes, int n_in,
                              void* d_out, int out_size, void* d_ws, size_t ws_size,
                              hipStream_t stream)
{
    const float* x            = (const float*)d_in[0];
    const float* W            = (const float*)d_in[1];
    const float* base_bias    = (const float*)d_in[2];
    const float* lora_a       = (const float*)d_in[3];
    const float* lora_b       = (const float*)d_in[4];
    const float* bias_stacked = (const float*)d_in[5];
    const int*   indices      = (const int*)d_in[6];
    float*       out          = (float*)d_out;

    const size_t sBytes  = (size_t)T_TOK * R_RANK * sizeof(float);   // 512 KB
    const size_t xbBytes = (size_t)T_TOK * D_IN * 2;                 // 64 MB
    const size_t wbBytes = (size_t)D_OUTD * D_IN * 2;                // 32 MB

    float* s = (float*)d_ws;

    // 1) LoRA shrink (always)
    lora_shrink_kernel<<<T_TOK, 256, 0, stream>>>(x, lora_a, indices, s);

    dim3 grid(D_OUTD / BN, T_TOK / BM);   // 32 x 64 blocks

    if (ws_size >= sBytes + xbBytes + wbBytes) {
        // 2) bf16 prepass into workspace
        __bf16* xb = (__bf16*)((char*)d_ws + sBytes);
        __bf16* wb = (__bf16*)((char*)d_ws + sBytes + xbBytes);
        const long xn4 = (long)T_TOK * D_IN / 4;
        const long wn4 = (long)D_OUTD * D_IN / 4;
        cvt_f32_to_bf16_kernel<<<(int)((xn4 + 255) / 256), 256, 0, stream>>>(x, xb, xn4);
        cvt_f32_to_bf16_kernel<<<(int)((wn4 + 255) / 256), 256, 0, stream>>>(W, wb, wn4);

        // 3) async double-buffered WMMA GEMM
        lora_gemm_async_kernel<<<grid, 256, 0, stream>>>(
            xb, wb, base_bias, lora_b, bias_stacked, indices, s, out);
    } else {
        lora_gemm_fallback_kernel<<<grid, 256, 0, stream>>>(
            x, W, base_bias, lora_b, bias_stacked, indices, s, out);
    }
}
